// ReBlockWOMeM_90340342104846
// MI455X (gfx1250) — compile-verified, matrix operand order checked by reading, not run on hardware
//
#include <hip/hip_runtime.h>
#include <math.h>

typedef __bf16 bf16_t;
typedef __attribute__((ext_vector_type(16))) __bf16 v16bf;
typedef __attribute__((ext_vector_type(8)))  float  v8f;

constexpr int kC  = 1024;
constexpr int kF  = 4096;
constexpr int kB  = 4;
constexpr int kT  = 2048;
constexpr int kH  = 16;
constexpr int kDh = 64;
constexpr int kM  = kB * kT;   // 8192 rows in the flattened (B*T, C) view

// ---------------------------------------------------------------------------
// Fragment helpers.
// 16-bit A-fragment layout (ISA 7.12.2): lane<16 holds K {0..7, 16..23},
// lane>=16 holds K {8..15, 24..31}; i.e. two contiguous 8-element (16B)
// chunks at offsets (kh*8) and (16 + kh*8). B fragments mirror this with
// lane = output column, so B^T (column-major-of-B) rows load identically.
// ---------------------------------------------------------------------------
__device__ __forceinline__ v16bf load_frag(const bf16_t* p0, const bf16_t* p1) {
  union { v16bf v; uint4 u[2]; } f;
  f.u[0] = *reinterpret_cast<const uint4*>(p0);
  f.u[1] = *reinterpret_cast<const uint4*>(p1);
  return f.v;
}

__device__ __forceinline__ v8f wmma_bf16(v16bf a, v16bf b, v8f c) {
  return __builtin_amdgcn_wmma_f32_16x16x32_bf16(
      /*neg_a=*/false, a, /*neg_b=*/false, b,
      /*c_mod=*/(short)0, c, /*reuse_a=*/false, /*reuse_b=*/false);
}

// ---------------------------------------------------------------------------
// Async global->LDS copy (16B per lane), ASYNCcnt-tracked.
// Builtin signature (from hipcc diagnostic): param0 = int4 addrspace(1)*
// (global src), param1 = int4 addrspace(3)* (LDS dst), imm offset, imm cpol.
// ---------------------------------------------------------------------------
#if defined(__has_builtin) && __has_builtin(__builtin_amdgcn_global_load_async_to_lds_b128)
#define ATHENA_HAS_ASYNC_LDS 1
#else
#define ATHENA_HAS_ASYNC_LDS 0
#endif

typedef int v4i_vs __attribute__((vector_size(16)));
typedef __attribute__((address_space(1))) v4i_vs g_v4i;
typedef __attribute__((address_space(3))) v4i_vs l_v4i;

__device__ __forceinline__ void async_cp16(void* lds_dst, const void* gsrc) {
#if ATHENA_HAS_ASYNC_LDS
  __builtin_amdgcn_global_load_async_to_lds_b128(
      (g_v4i*)gsrc, (l_v4i*)lds_dst, 0, 0);
#else
  *reinterpret_cast<uint4*>(lds_dst) = *reinterpret_cast<const uint4*>(gsrc);
#endif
}

__device__ __forceinline__ void wait_async0() {
#if defined(__has_builtin) && __has_builtin(__builtin_amdgcn_s_wait_asynccnt)
  __builtin_amdgcn_s_wait_asynccnt(0);
#else
  asm volatile("s_wait_asynccnt 0" ::: "memory");
#endif
}

// ---------------------------------------------------------------------------
// fp32 [K,N] weight -> bf16 [N,K] (transposed) for WMMA B fragments.
// ---------------------------------------------------------------------------
__global__ void k_convT(const float* __restrict__ W, bf16_t* __restrict__ Wt,
                        int K, int N) {
  long long i = (long long)blockIdx.x * blockDim.x + threadIdx.x;
  if (i >= (long long)K * N) return;
  const int n = (int)(i % N);
  const int k = (int)(i / N);
  Wt[(long long)n * K + k] = (bf16_t)W[i];
}

// ---------------------------------------------------------------------------
// LayerNorm over C=1024, one block per row, bf16 output (WMMA A operand).
// ---------------------------------------------------------------------------
__global__ void k_layernorm(const float* __restrict__ x, const float* __restrict__ g,
                            const float* __restrict__ be, bf16_t* __restrict__ y,
                            int C) {
  __shared__ float red[256];
  const int row = blockIdx.x;
  const int tid = threadIdx.x;
  const float* xr = x + (long long)row * C;
  float s = 0.f;
  for (int i = tid; i < C; i += 256) s += xr[i];
  red[tid] = s; __syncthreads();
  for (int off = 128; off > 0; off >>= 1) {
    if (tid < off) red[tid] += red[tid + off];
    __syncthreads();
  }
  const float mu = red[0] * (1.0f / C);
  __syncthreads();
  float vs = 0.f;
  for (int i = tid; i < C; i += 256) { float d = xr[i] - mu; vs += d * d; }
  red[tid] = vs; __syncthreads();
  for (int off = 128; off > 0; off >>= 1) {
    if (tid < off) red[tid] += red[tid + off];
    __syncthreads();
  }
  const float rstd = rsqrtf(red[0] * (1.0f / C) + 1e-5f);
  bf16_t* yr = y + (long long)row * C;
  for (int i = tid; i < C; i += 256)
    yr[i] = (bf16_t)((xr[i] - mu) * rstd * g[i] + be[i]);
}

// ---------------------------------------------------------------------------
// Cooperative GEMM: out[M,N] = A[M,K](bf16) @ B[K,N] + bias, with B given
// transposed bf16 [N,K]. Block = 256 threads (8 waves) computing a 128x128
// tile; per 32-wide K step the block double-buffers a 128x32 A slab and a
// 128x32 B^T slab into LDS with async global->LDS copies, then each wave
// (arranged 4x2) runs 8 WMMAs on a 32x64 register tile. Optional exact
// GELU, fp32 residual add, fp32/bf16 stores.
// ---------------------------------------------------------------------------
__global__ void __launch_bounds__(256)
k_gemm(const bf16_t* __restrict__ A, const bf16_t* __restrict__ Bt,
       const float* __restrict__ bias, const float* __restrict__ res,
       float* __restrict__ outF, bf16_t* __restrict__ outB,
       int M, int N, int K, int doGelu) {
  __shared__ __align__(16) bf16_t sA[2][128][32];
  __shared__ __align__(16) bf16_t sB[2][128][32];

  const int tid  = threadIdx.x;
  const int lane = tid & 31;
  const int w    = tid >> 5;           // 0..7
  const int nT   = N >> 7;
  const int mi   = blockIdx.x / nT, ni = blockIdx.x % nT;
  if (mi * 128 >= M) return;
  const int m0 = mi << 7, n0 = ni << 7;
  const int wm = w >> 1;               // 0..3  (32-row band)
  const int wn = w & 1;                // 0..1  (64-col band)
  const int r  = lane & 15, kh = lane >> 4;

  const bf16_t* Ag = A  + (long long)m0 * K;   // 128 rows
  const bf16_t* Bg = Bt + (long long)n0 * K;   // 128 rows of B^T

  // staging map: thread -> (row = tid>>1, 32B half-row = tid&1), 4 x 16B
  const int srow = tid >> 1;
  const int scol = (tid & 1) * 16;

  auto stage = [&](int k, int buf) {
    const bf16_t* ga = Ag + (long long)srow * K + k + scol;
    async_cp16(&sA[buf][srow][scol], ga);
    async_cp16(&sA[buf][srow][scol + 8], ga + 8);
    const bf16_t* gb = Bg + (long long)srow * K + k + scol;
    async_cp16(&sB[buf][srow][scol], gb);
    async_cp16(&sB[buf][srow][scol + 8], gb + 8);
  };

  v8f acc[2][4] = {};
  stage(0, 0);
  for (int k = 0, it = 0; k < K; k += 32, ++it) {
    const int cur = it & 1;
    wait_async0();
    __syncthreads();                       // staged slabs visible to all waves
    if (k + 32 < K) stage(k + 32, cur ^ 1);
    v16bf a0, a1;
    {
      const bf16_t* ap0 = &sA[cur][wm * 32 + r][kh * 8];
      const bf16_t* ap1 = &sA[cur][wm * 32 + 16 + r][kh * 8];
      a0 = load_frag(ap0, ap0 + 16);
      a1 = load_frag(ap1, ap1 + 16);
    }
#pragma unroll
    for (int t = 0; t < 4; ++t) {
      const bf16_t* bp = &sB[cur][wn * 64 + t * 16 + r][kh * 8];
      v16bf bb = load_frag(bp, bp + 16);
      acc[0][t] = wmma_bf16(a0, bb, acc[0][t]);
      acc[1][t] = wmma_bf16(a1, bb, acc[1][t]);
    }
    __syncthreads();                       // done reading before re-staging
  }

  // C/D layout: VGPR rr -> row (+ rr + 8*kh), col = t*16 + (lane&15)
#pragma unroll
  for (int rb = 0; rb < 2; ++rb) {
#pragma unroll
    for (int t = 0; t < 4; ++t) {
      const int col = n0 + wn * 64 + t * 16 + r;
      const float bc = bias[col];
#pragma unroll
      for (int rr = 0; rr < 8; ++rr) {
        const int row = m0 + wm * 32 + rb * 16 + rr + 8 * kh;
        float v = acc[rb][t][rr] + bc;
        if (doGelu) v = 0.5f * v * (1.0f + erff(v * 0.70710678118654752f));
        const long long idx = (long long)row * N + col;
        if (res)  v += res[idx];
        if (outF) outF[idx] = v;
        if (outB) outB[idx] = (bf16_t)v;
      }
    }
  }
}

// ---------------------------------------------------------------------------
// V (bf16 [B*T, C], head-interleaved) -> Vt (bf16 [B,H,Dh,T]) so PV
// B-fragments are contiguous along the key axis.
// ---------------------------------------------------------------------------
__global__ void k_transV(const bf16_t* __restrict__ V, bf16_t* __restrict__ Vt) {
  long long i = (long long)blockIdx.x * blockDim.x + threadIdx.x;
  if (i >= (long long)kM * kC) return;
  const int dh = (int)(i & (kDh - 1));
  long long j = i >> 6;                 // / Dh
  const int h = (int)(j & (kH - 1));
  j >>= 4;                              // / H
  const int t = (int)(j & (kT - 1));
  const int b = (int)(j >> 11);         // / T
  Vt[(((long long)(b * kH + h)) * kDh + dh) * kT + t] = V[i];
}

// ---------------------------------------------------------------------------
// Flash attention: one wave per (b, h, 16-query block). Per 32-key block:
// 4 WMMAs for Q·K^T (dh split into 2 k-steps x 2 key tiles), causal mask +
// online softmax through wave-private LDS, then 4 WMMAs for P·V.
// ---------------------------------------------------------------------------
__global__ void __launch_bounds__(128)
k_attn(const bf16_t* __restrict__ Q, const bf16_t* __restrict__ Kmat,
       const bf16_t* __restrict__ Vt, bf16_t* __restrict__ Y) {
  __shared__ __align__(16) float  Sbuf[4][16][32];
  __shared__ __align__(16) bf16_t Pbuf[4][16][32];
  __shared__ float mrow[4][16], lrow[4][16], srow[4][16];

  const int lane = threadIdx.x & 31;
  const int w    = threadIdx.x >> 5;
  const int gw   = blockIdx.x * 4 + w;
  const int nqb  = kT / 16;
  if (gw >= kB * kH * nqb) return;
  const int qb = gw % nqb;
  const int h  = (gw / nqb) % kH;
  const int b  = gw / (nqb * kH);
  const int q0 = qb * 16;
  const int r  = lane & 15, kh = lane >> 4;

  const bf16_t* Qb = Q    + (long long)b * kT * kC + (long long)h * kDh;
  const bf16_t* Kb = Kmat + (long long)b * kT * kC + (long long)h * kDh;
  const bf16_t* Vb = Vt   + (long long)(b * kH + h) * kDh * kT;

  v16bf aq[2];
#pragma unroll
  for (int s = 0; s < 2; ++s) {
    const bf16_t* p = Qb + (long long)(q0 + r) * kC + 32 * s + kh * 8;
    aq[s] = load_frag(p, p + 16);
  }
  v8f o[4] = {};
  if (lane < 16) { mrow[w][lane] = -3.0e38f; lrow[w][lane] = 0.f; }

  for (int kb = 0; kb <= q0 + 15; kb += 32) {
    // --- S = Q . K^T for 16 queries x 32 keys ---
    v8f s0 = {}, s1 = {};
#pragma unroll
    for (int s = 0; s < 2; ++s) {
      const bf16_t* p0 = Kb + (long long)(kb + r) * kC + 32 * s + kh * 8;
      const bf16_t* p1 = Kb + (long long)(kb + 16 + r) * kC + 32 * s + kh * 8;
      v16bf b0 = load_frag(p0, p0 + 16);
      v16bf b1 = load_frag(p1, p1 + 16);
      s0 = wmma_bf16(aq[s], b0, s0);
      s1 = wmma_bf16(aq[s], b1, s1);
    }
    // --- scale + causal mask into LDS ---
#pragma unroll
    for (int rr = 0; rr < 8; ++rr) {
      const int ql = rr + 8 * kh;
      const int qg = q0 + ql;
      Sbuf[w][ql][r]      = (kb + r      <= qg) ? s0[rr] * 0.125f : -3.0e38f;
      Sbuf[w][ql][16 + r] = (kb + 16 + r <= qg) ? s1[rr] * 0.125f : -3.0e38f;
    }
    // --- online softmax, lanes 0..15 each own one query row ---
    if (lane < 16) {
      const int q = lane;
      const float mold = mrow[w][q];
      float mx = mold;
#pragma unroll
      for (int j = 0; j < 32; ++j) mx = fmaxf(mx, Sbuf[w][q][j]);
      const float sc = __expf(mold - mx);
      float sum = 0.f;
#pragma unroll
      for (int j = 0; j < 32; ++j) {
        const float p = __expf(Sbuf[w][q][j] - mx);
        sum += p;
        Pbuf[w][q][j] = (bf16_t)p;
      }
      mrow[w][q] = mx;
      lrow[w][q] = lrow[w][q] * sc + sum;
      srow[w][q] = sc;
    }
    // --- rescale running output ---
    float sc8[8];
#pragma unroll
    for (int rr = 0; rr < 8; ++rr) sc8[rr] = srow[w][rr + 8 * kh];
#pragma unroll
    for (int t = 0; t < 4; ++t)
#pragma unroll
      for (int rr = 0; rr < 8; ++rr) o[t][rr] *= sc8[rr];
    // --- O += P . V ---
    const bf16_t* pp = &Pbuf[w][r][kh * 8];
    v16bf ap = load_frag(pp, pp + 16);
#pragma unroll
    for (int t = 0; t < 4; ++t) {
      const bf16_t* pv = Vb + (long long)(t * 16 + r) * kT + kb + kh * 8;
      v16bf bv = load_frag(pv, pv + 16);
      o[t] = wmma_bf16(ap, bv, o[t]);
    }
  }
  // --- normalize and emit bf16 Y (input to output projection GEMM) ---
  float inv8[8];
#pragma unroll
  for (int rr = 0; rr < 8; ++rr) inv8[rr] = 1.0f / lrow[w][rr + 8 * kh];
#pragma unroll
  for (int t = 0; t < 4; ++t)
#pragma unroll
    for (int rr = 0; rr < 8; ++rr) {
      const int row = q0 + rr + 8 * kh;
      const int col = h * kDh + t * 16 + r;
      Y[((long long)b * kT + row) * kC + col] = (bf16_t)(o[t][rr] * inv8[rr]);
    }
}

// ---------------------------------------------------------------------------
// Host orchestration.
// ---------------------------------------------------------------------------
extern "C" void kernel_launch(void* const* d_in, const int* in_sizes, int n_in,
                              void* d_out, int out_size, void* d_ws, size_t ws_size,
                              hipStream_t stream) {
  (void)in_sizes; (void)n_in; (void)out_size; (void)ws_size;
  const float* x0 = (const float*)d_in[0];
  const float* a1w[4] = {(const float*)d_in[1],  (const float*)d_in[3],
                         (const float*)d_in[5],  (const float*)d_in[7]};
  const float* a1b[4] = {(const float*)d_in[2],  (const float*)d_in[4],
                         (const float*)d_in[6],  (const float*)d_in[8]};
  const float* a2w[4] = {(const float*)d_in[9],  (const float*)d_in[11],
                         (const float*)d_in[13], (const float*)d_in[15]};
  const float* a2b[4] = {(const float*)d_in[10], (const float*)d_in[12],
                         (const float*)d_in[14], (const float*)d_in[16]};
  const float* m1w1 = (const float*)d_in[17]; const float* m1b1 = (const float*)d_in[18];
  const float* m1w2 = (const float*)d_in[19]; const float* m1b2 = (const float*)d_in[20];
  const float* m2w1 = (const float*)d_in[21]; const float* m2b1 = (const float*)d_in[22];
  const float* m2w2 = (const float*)d_in[23]; const float* m2b2 = (const float*)d_in[24];
  const float* lng[4] = {(const float*)d_in[25], (const float*)d_in[27],
                         (const float*)d_in[29], (const float*)d_in[31]};
  const float* lnb[4] = {(const float*)d_in[26], (const float*)d_in[28],
                         (const float*)d_in[30], (const float*)d_in[32]};

  char* ws = (char*)d_ws;
  const size_t MB = 1024ull * 1024ull;
  bf16_t* WqT = (bf16_t*)(ws + 0 * MB);     // [C,C] bf16, transposed
  bf16_t* WkT = (bf16_t*)(ws + 2 * MB);
  bf16_t* WvT = (bf16_t*)(ws + 4 * MB);
  bf16_t* WoT = (bf16_t*)(ws + 6 * MB);
  bf16_t* W1T = (bf16_t*)(ws + 8 * MB);     // [F,C] bf16 (w1 transposed)
  bf16_t* W2T = (bf16_t*)(ws + 16 * MB);    // [C,F] bf16 (w2 transposed)
  bf16_t* XB  = (bf16_t*)(ws + 24 * MB);    // LN output, bf16 [M,C]
  bf16_t* QB  = (bf16_t*)(ws + 40 * MB);    // bf16 [M,C]
  bf16_t* KB  = (bf16_t*)(ws + 56 * MB);
  bf16_t* VB  = (bf16_t*)(ws + 72 * MB);    // also reused as attention Y
  bf16_t* VT  = (bf16_t*)(ws + 88 * MB);    // [B,H,Dh,T]
  bf16_t* GB  = QB;                         // MLP hidden bf16 [M,F] = 64 MB span
  float*  XA  = (float*)(ws + 104 * MB);    // residual stream scratch fp32 [M,C]
  float*  XOUT = (float*)d_out;

  auto convT = [&](const float* W, bf16_t* Wt, int K, int N) {
    const long long tot = (long long)K * N;
    k_convT<<<(int)((tot + 255) / 256), 256, 0, stream>>>(W, Wt, K, N);
  };
  auto gemm = [&](const bf16_t* A, const bf16_t* Bt, const float* bias,
                  const float* res, float* oF, bf16_t* oB, int N, int K, int gelu) {
    const int blocks = (kM / 128) * (N / 128);
    k_gemm<<<blocks, 256, 0, stream>>>(A, Bt, bias, res, oF, oB, kM, N, K, gelu);
  };

  auto layer = [&](const float* xin, float* xout,
                   const float* const* aw, const float* const* ab,
                   const float* w1, const float* b1,
                   const float* w2, const float* b2,
                   const float* ga, const float* ba,
                   const float* gm, const float* bm) {
    convT(aw[0], WqT, kC, kC);
    convT(aw[1], WkT, kC, kC);
    convT(aw[2], WvT, kC, kC);
    convT(aw[3], WoT, kC, kC);
    convT(w1, W1T, kC, kF);
    convT(w2, W2T, kF, kC);
    k_layernorm<<<kM, 256, 0, stream>>>(xin, ga, ba, XB, kC);
    gemm(XB, WqT, ab[0], nullptr, nullptr, QB, kC, kC, 0);
    gemm(XB, WkT, ab[1], nullptr, nullptr, KB, kC, kC, 0);
    gemm(XB, WvT, ab[2], nullptr, nullptr, VB, kC, kC, 0);
    k_transV<<<(int)(((long long)kM * kC + 255) / 256), 256, 0, stream>>>(VB, VT);
    k_attn<<<(kB * kH * (kT / 16)) / 4, 128, 0, stream>>>(QB, KB, VT, VB);
    gemm(VB, WoT, ab[3], xin, XA, nullptr, kC, kC, 0);          // x += attn out
    k_layernorm<<<kM, 256, 0, stream>>>(XA, gm, bm, XB, kC);
    gemm(XB, W1T, b1, nullptr, nullptr, GB, kF, kC, 1);         // gelu hidden
    gemm(GB, W2T, b2, XA, xout, nullptr, kC, kF, 0);            // x += mlp out
  };

  layer(x0,   XOUT, a1w, a1b, m1w1, m1b1, m1w2, m1b2, lng[0], lnb[0], lng[1], lnb[1]);
  layer(XOUT, XOUT, a2w, a2b, m2w1, m2b1, m2w2, m2b2, lng[2], lnb[2], lng[3], lnb[3]);
}
